// SqueezeFRCN_8297876815890
// MI455X (gfx1250) — matrix operand order repaired and checked by reading, not compiled, not run on hardware
//
#include <hip/hip_runtime.h>
#include <hip/hip_bf16.h>
#include <cstdint>
#include <cstddef>

// ---------------------------------------------------------------------------
// RPN head for SqueezeFRCN on gfx1250 (MI455X): bf16 WMMA implicit-GEMM conv
// (double-buffered LDS, TDM weight tiles), WMMA 1x1 head GEMM, decode + NMS.
// ---------------------------------------------------------------------------

typedef __attribute__((ext_vector_type(16))) __bf16 v16bf;
typedef __attribute__((ext_vector_type(8)))  __bf16 v8bf;
typedef __attribute__((ext_vector_type(8)))  float  v8f;

#define NIMG   16
#define CIN    256
#define HDIM   64
#define WDIM   64
#define MIDCH  256
#define NANCH  9
#define KA     (HDIM * WDIM * NANCH)   // 36864 anchors / image
#define MROWS  (NIMG * HDIM * WDIM)    // 65536 GEMM rows
#define KCONV  (CIN * 9)               // 2304
#define N2     64                      // loc(36)+score(18)+pad(10)
#define POSTN  300

// d_out layout (float elements), concatenated reference outputs:
#define LOC_OFF   ((size_t)0)
#define SCORE_OFF ((size_t)NIMG * KA * 4)                    // 2359296
#define ROIS_OFF  (SCORE_OFF + (size_t)NIMG * KA * 2)        // 3538944
#define RIDX_OFF  (ROIS_OFF + (size_t)NIMG * POSTN * 4)      // 3558144
#define ANCH_OFF  (RIDX_OFF + (size_t)NIMG * POSTN)          // 3562944

#if defined(__has_builtin)
#if __has_builtin(__builtin_amdgcn_tensor_load_to_lds)
#define HAVE_TDM 1
#endif
#endif
#ifndef HAVE_TDM
#define HAVE_TDM 0
#endif

static __device__ __forceinline__ __bf16 f2bf(float f) {
  unsigned u = __builtin_bit_cast(unsigned, f);
  u += 0x7FFFu + ((u >> 16) & 1u);               // round-nearest-even
  unsigned short h = (unsigned short)(u >> 16);
  return __builtin_bit_cast(__bf16, h);
}

static __device__ __forceinline__ v16bf ldfrag(const __bf16* p) {
  // 32B contiguous LDS read -> 2x ds_load_b128, concatenated to a v16bf frag
  v8bf lo = ((const v8bf*)p)[0];
  v8bf hi = ((const v8bf*)p)[1];
  return __builtin_shufflevector(lo, hi, 0, 1, 2, 3, 4, 5, 6, 7,
                                 8, 9, 10, 11, 12, 13, 14, 15);
}

#if HAVE_TDM
typedef unsigned int v4u __attribute__((ext_vector_type(4)));
typedef int v4i __attribute__((ext_vector_type(4)));
typedef int v8i __attribute__((ext_vector_type(8)));

// TDM: 2D weight tile 64 rows x 32 cols (bf16), row stride 2304 elements in
// memory, LDS padded to a 40-element (80B) row stride via pad_enable:
// pad_interval=3 (every 16 DWORDs = one 32-col row) , pad_amount=3 (4 DWORDs).
static __device__ __forceinline__ void tdm_load_b_tile(const __bf16* gsrc,
                                                       unsigned lds_off) {
  unsigned long long ga = (unsigned long long)(size_t)(const void*)gsrc;
  v4u g0;
  g0.x = 1u;                                                // count=1, user D#
  g0.y = lds_off;                                           // lds_addr (bytes)
  g0.z = (unsigned)ga;                                      // global_addr[31:0]
  g0.w = (unsigned)((ga >> 32) & 0x01FFFFFFu) | (2u << 30); // [56:32] | type=2
  const unsigned td0 = (unsigned)KCONV;   // tensor_dim0
  const unsigned td1 = 256u;              // tensor_dim1
  const unsigned ti0 = 32u;               // tile_dim0 (K chunk)
  const unsigned ti1 = 64u;               // tile_dim1 (out-channel rows)
  const unsigned long long st0 = (unsigned long long)KCONV;
  v8i g1;
  g1[0] = (int)((1u << 16) | (1u << 20) | (3u << 22) | (3u << 25));
  g1[1] = (int)((td0 & 0xFFFFu) << 16);
  g1[2] = (int)((td0 >> 16) | ((td1 & 0xFFFFu) << 16));
  g1[3] = (int)((td1 >> 16) | (ti0 << 16));
  g1[4] = (int)ti1;                       // tile_dim1 | tile_dim2=0
  g1[5] = (int)(unsigned)(st0 & 0xFFFFFFFFull);
  g1[6] = (int)(unsigned)(st0 >> 32);     // stride0[47:32] | stride1[15:0]=0
  g1[7] = 0;
  v4i gz = {0, 0, 0, 0};
#if defined(__clang_major__) && (__clang_major__ >= 23)
  v8i gz8 = {0, 0, 0, 0, 0, 0, 0, 0};
  __builtin_amdgcn_tensor_load_to_lds(g0, g1, gz, gz, gz8, 0);
#else
  __builtin_amdgcn_tensor_load_to_lds(g0, g1, gz, gz, 0);
#endif
}
#endif  // HAVE_TDM

// ---------------------------------------------------------------------------
// Prep kernels: layout transforms + fp32 -> bf16
// ---------------------------------------------------------------------------

// x: NCHW f32 -> xb: NHWC bf16
__global__ __launch_bounds__(256) void prep_x(const float* __restrict__ x,
                                              __bf16* __restrict__ xb) {
  size_t i = (size_t)blockIdx.x * 256 + threadIdx.x;   // dest index
  int c    = (int)(i & 255);
  size_t r = i >> 8;                                   // n*4096 + pix
  int pix  = (int)(r & 4095);
  int n    = (int)(r >> 12);
  xb[i] = f2bf(x[(((size_t)n * 256 + c) * 4096) + pix]);
}

// conv1_w: OIHW(256,256,3,3) f32 -> wt: [O][k] bf16, k = tap*256 + c
__global__ __launch_bounds__(256) void prep_w1(const float* __restrict__ w,
                                               __bf16* __restrict__ wt) {
  size_t i = (size_t)blockIdx.x * 256 + threadIdx.x;   // 0 .. 589823
  int o = (int)(i / KCONV);
  int k = (int)(i - (size_t)o * KCONV);
  int t = k >> 8;           // tap 0..8
  int c = k & 255;
  wt[i] = f2bf(w[((size_t)o * 256 + c) * 9 + t]);
}

// loc_w(36,256) + score_w(18,256) -> w2t: [64][256] bf16 (rows 54..63 zero)
__global__ __launch_bounds__(256) void prep_w2(const float* __restrict__ loc_w,
                                               const float* __restrict__ score_w,
                                               const float* __restrict__ loc_b,
                                               const float* __restrict__ score_b,
                                               __bf16* __restrict__ w2t,
                                               float* __restrict__ b2) {
  int i = blockIdx.x * 256 + threadIdx.x;   // 0..16383
  int o = i >> 8;
  int k = i & 255;
  float v = 0.0f;
  if (o < 36)       v = loc_w[(size_t)o * 256 + k];
  else if (o < 54)  v = score_w[(size_t)(o - 36) * 256 + k];
  w2t[i] = f2bf(v);
  if (k == 0) {
    float b = 0.0f;
    if (o < 36)      b = loc_b[o];
    else if (o < 54) b = score_b[o - 36];
    b2[o] = b;
  }
}

// ---------------------------------------------------------------------------
// 3x3 conv as implicit GEMM: D[65536,256] = Patch[65536,2304] x W[2304,256]
// Block = 256 threads = 8 wave32s; tile 128(M) x 64(N); K step = 32.
// Double-buffered LDS; B tiles via Tensor Data Mover when available.
// ---------------------------------------------------------------------------
__global__ __launch_bounds__(256) void conv3x3_wmma(
    const __bf16* __restrict__ xb,    // [N][64][64][256]
    const __bf16* __restrict__ wt,    // [256][2304]
    const float*  __restrict__ bias,  // [256]
    __bf16*       __restrict__ hout)  // [65536][256]
{
  __shared__ __bf16 As[2][128 * 40];   // contiguous K layout, stride 40
  __shared__ __bf16 Bs[2][64 * 40];

  const int tid   = threadIdx.x;
  const int tileM = blockIdx.x * 128;
  const int tileN = blockIdx.y * 64;
  const int lane  = tid & 31;
  const int wave  = tid >> 5;
  const int waveM = wave & 3;        // 4 waves along M (32 rows each)
  const int waveN = wave >> 2;       // 2 waves along N (32 cols each)

  // A global-load mapping: 128 rows x 32 cols per step; thread = (row, half)
  const int arow  = tid >> 1;        // 0..127
  const int ahalf = tid & 1;         // 16-channel half
  const int m     = tileM + arow;
  const int pix   = m & 4095;
  const int py    = pix >> 6;
  const int px    = pix & 63;
  const __bf16* abase = xb + (((size_t)m) << 8) + (ahalf << 4);

  // B fallback global-load mapping
  const int brow = tid >> 2;         // 0..63
  const int bq   = tid & 3;

  // fragment lane decomposition
  const int fr = lane & 15;
  const int kh = lane >> 4;

  v8f acc[2][2] = {};
  const v8bf vz = {};

  v8bf a0 = vz, a1 = vz;
#if !HAVE_TDM
  v8bf bv = vz;
#endif

  auto loadA = [&](int ks, v8bf& r0, v8bf& r1) {
    const int tap = ks >> 3;
    const int c0  = (ks & 7) << 5;
    const int t3  = tap / 3;
    const int dy  = t3 - 1;
    const int dx  = tap - t3 * 3 - 1;
    const int yy  = py + dy;
    const int xx  = px + dx;
    r0 = vz;
    r1 = vz;
    if ((unsigned)yy < (unsigned)HDIM && (unsigned)xx < (unsigned)WDIM) {
      const __bf16* s = abase + ((((dy << 6) + dx) << 8) + c0);
      r0 = ((const v8bf*)s)[0];
      r1 = ((const v8bf*)s)[1];
      __builtin_prefetch(s + 32, 0, 1);            // global_prefetch_b8
    }
  };
  auto storeA = [&](int buf, v8bf r0, v8bf r1) {
    // contiguous k: j0..15 -> half*16 + j (matches B layout)
    __bf16* d = &As[buf][arow * 40];
    ((v8bf*)(d + (ahalf << 4)))[0]      = r0;
    ((v8bf*)(d + (ahalf << 4) + 8))[0]  = r1;
  };
#if HAVE_TDM
  auto issueB = [&](int ks, int buf) {
    if (wave == 0) {
      const __bf16* gsrc = wt + (size_t)tileN * KCONV + (ks << 5);
      unsigned lds_off = (unsigned)(size_t)(void*)&Bs[buf][0];
      tdm_load_b_tile(gsrc, lds_off);
    }
  };
#else
  auto loadB = [&](int ks) {
    return *(const v8bf*)(wt + (size_t)(tileN + brow) * KCONV + (ks << 5) + (bq << 3));
  };
  auto storeB = [&](int buf, v8bf v) {
    ((v8bf*)(&Bs[buf][brow * 40 + (bq << 3)]))[0] = v;
  };
#endif

  // prologue: stage tile 0
  loadA(0, a0, a1);
#if HAVE_TDM
  issueB(0, 0);
#else
  bv = loadB(0);
#endif
  storeA(0, a0, a1);
#if !HAVE_TDM
  storeB(0, bv);
#endif
#if HAVE_TDM
  __builtin_amdgcn_s_wait_tensorcnt(0);
#endif
  __syncthreads();

  const int KSTEPS = KCONV / 32;                       // 72
  for (int ks = 0; ks < KSTEPS; ++ks) {
    const int cur  = ks & 1;
    const int nb   = cur ^ 1;
    const bool more = (ks + 1) < KSTEPS;

    if (more) {                                        // prefetch next tile
      loadA(ks + 1, a0, a1);
#if HAVE_TDM
      issueB(ks + 1, nb);
#else
      bv = loadB(ks + 1);
#endif
    }

    // compute current tile from LDS
    v16bf af[2], bf[2];
#pragma unroll
    for (int i = 0; i < 2; ++i)
      af[i] = ldfrag(&As[cur][(waveM * 32 + i * 16 + fr) * 40 + kh * 16]);
#pragma unroll
    for (int j = 0; j < 2; ++j)
      bf[j] = ldfrag(&Bs[cur][(waveN * 32 + j * 16 + fr) * 40 + kh * 16]);

#pragma unroll
    for (int i = 0; i < 2; ++i)
#pragma unroll
      for (int j = 0; j < 2; ++j)
        acc[i][j] = __builtin_amdgcn_wmma_f32_16x16x32_bf16(
            false, af[i], false, bf[j], (short)0, acc[i][j], false, false);

    if (more) {
      storeA(nb, a0, a1);
#if !HAVE_TDM
      storeB(nb, bv);
#endif
    }
#if HAVE_TDM
    __builtin_amdgcn_s_wait_tensorcnt(0);
#endif
    __syncthreads();
  }

  // epilogue: +bias, ReLU, bf16 store for second GEMM
#pragma unroll
  for (int i = 0; i < 2; ++i) {
#pragma unroll
    for (int j = 0; j < 2; ++j) {
      const int col   = tileN + waveN * 32 + j * 16 + fr;
      const float b   = bias[col];
      const int mbase = tileM + waveM * 32 + i * 16 + kh * 8;
#pragma unroll
      for (int r = 0; r < 8; ++r) {
        float v = acc[i][j][r] + b;
        v = v > 0.0f ? v : 0.0f;
        hout[(size_t)(mbase + r) * 256 + col] = f2bf(v);
      }
    }
  }
}

// ---------------------------------------------------------------------------
// Fused 1x1 loc+score GEMM: out2[65536,64] = h[65536,256] x w2t^T[256,64]
// ---------------------------------------------------------------------------
__global__ __launch_bounds__(256) void gemm2_wmma(
    const __bf16* __restrict__ hA,    // [65536][256]
    const __bf16* __restrict__ w2t,   // [64][256]
    const float*  __restrict__ b2,    // [64]
    float*        __restrict__ out2)  // [65536][64]
{
  __shared__ __bf16 As[128 * 40];
  __shared__ __bf16 Bs[64 * 40];

  const int tid   = threadIdx.x;
  const int tileM = blockIdx.x * 128;
  const int lane  = tid & 31;
  const int wave  = tid >> 5;
  const int waveM = wave & 3;
  const int waveN = wave >> 2;

  const int arow  = tid >> 1;
  const int ahalf = tid & 1;
  const int brow  = tid >> 2;
  const int bq    = tid & 3;
  const int fr    = lane & 15;
  const int kh    = lane >> 4;

  v8f acc[2][2] = {};

  for (int ks = 0; ks < 256 / 32; ++ks) {             // 8 steps
    {
      const __bf16* src = hA + (size_t)(tileM + arow) * 256 + (ks << 5) + (ahalf << 4);
      v8bf a0 = ((const v8bf*)src)[0];
      v8bf a1 = ((const v8bf*)src)[1];
      __bf16* d = As + arow * 40;
      ((v8bf*)(d + (ahalf << 4)))[0]      = a0;
      ((v8bf*)(d + (ahalf << 4) + 8))[0]  = a1;
    }
    {
      const __bf16* bsrc = w2t + (size_t)brow * 256 + (ks << 5) + (bq << 3);
      ((v8bf*)(Bs + brow * 40 + (bq << 3)))[0] = *(const v8bf*)bsrc;
    }
    __syncthreads();

    v16bf af[2], bf[2];
#pragma unroll
    for (int i = 0; i < 2; ++i)
      af[i] = ldfrag(As + (waveM * 32 + i * 16 + fr) * 40 + kh * 16);
#pragma unroll
    for (int j = 0; j < 2; ++j)
      bf[j] = ldfrag(Bs + (waveN * 32 + j * 16 + fr) * 40 + kh * 16);

#pragma unroll
    for (int i = 0; i < 2; ++i)
#pragma unroll
      for (int j = 0; j < 2; ++j)
        acc[i][j] = __builtin_amdgcn_wmma_f32_16x16x32_bf16(
            false, af[i], false, bf[j], (short)0, acc[i][j], false, false);

    __syncthreads();
  }

#pragma unroll
  for (int i = 0; i < 2; ++i) {
#pragma unroll
    for (int j = 0; j < 2; ++j) {
      const int col   = waveN * 32 + j * 16 + fr;
      const float b   = b2[col];
      const int mbase = tileM + waveM * 32 + i * 16 + kh * 8;
#pragma unroll
      for (int r = 0; r < 8; ++r)
        out2[(size_t)(mbase + r) * N2 + col] = acc[i][j][r] + b;
    }
  }
}

// ---------------------------------------------------------------------------
// Decode: anchors + box transform + clip + min-size filter.
// ---------------------------------------------------------------------------
__global__ __launch_bounds__(256) void decode_kernel(
    const float* __restrict__ out2, const int* __restrict__ img_h_p,
    const int* __restrict__ img_w_p, float* __restrict__ dout,
    float* __restrict__ scores, float* __restrict__ boxes)
{
  int i = blockIdx.x * 256 + threadIdx.x;   // 0 .. 16*36864-1
  if (i >= NIMG * KA) return;
  int n   = i / KA;
  int pa  = i - n * KA;                     // pix*9 + a
  int pix = pa / NANCH;
  int a   = pa - pix * NANCH;
  int py  = pix >> 6;
  int px  = pix & 63;
  int m   = n * 4096 + pix;

  const float* o = out2 + (size_t)m * N2;
  float l0 = o[a * 4 + 0], l1 = o[a * 4 + 1];
  float l2 = o[a * 4 + 2], l3 = o[a * 4 + 3];
  float s0 = o[36 + a * 2 + 0];
  float s1 = o[36 + a * 2 + 1];

  const float srt[3] = {0.70710678f, 1.0f, 1.41421356f};   // sqrt(ratio)
  const float scl[3] = {8.0f, 16.0f, 32.0f};
  int ri = a / 3, si = a - ri * 3;
  float ah  = 16.0f * scl[si] * srt[ri];
  float aw  = 16.0f * scl[si] / srt[ri];
  float acy = (float)py * 32.0f + 8.0f;
  float acx = (float)px * 32.0f + 8.0f;

  float cy = l0 * ah + acy;
  float cx = l1 * aw + acx;
  float bh = __expf(l2) * ah;
  float bw = __expf(l3) * aw;
  float imh = (float)img_h_p[0];
  float imw = (float)img_w_p[0];
  float b0 = fminf(fmaxf(cy - 0.5f * bh, 0.0f), imh);
  float b1 = fminf(fmaxf(cx - 0.5f * bw, 0.0f), imw);
  float b2 = fminf(fmaxf(cy + 0.5f * bh, 0.0f), imh);
  float b3 = fminf(fmaxf(cx + 0.5f * bw, 0.0f), imw);
  float hs = b2 - b0, wsz = b3 - b1;
  float sv = (hs >= 16.0f && wsz >= 16.0f) ? s1 : -__builtin_inff();

  size_t li = (size_t)i * 4;
  dout[LOC_OFF + li + 0] = l0;
  dout[LOC_OFF + li + 1] = l1;
  dout[LOC_OFF + li + 2] = l2;
  dout[LOC_OFF + li + 3] = l3;
  dout[SCORE_OFF + (size_t)i * 2 + 0] = s0;
  dout[SCORE_OFF + (size_t)i * 2 + 1] = s1;
  scores[i] = sv;
  boxes[li + 0] = b0;
  boxes[li + 1] = b1;
  boxes[li + 2] = b2;
  boxes[li + 3] = b3;
  if (n == 0) {
    size_t ai = ANCH_OFF + (size_t)pa * 4;
    dout[ai + 0] = acy - 0.5f * ah;
    dout[ai + 1] = acx - 0.5f * aw;
    dout[ai + 2] = acy + 0.5f * ah;
    dout[ai + 3] = acx + 0.5f * aw;
  }
}

// ---------------------------------------------------------------------------
// NMS: one block / image; 300 iterations of block argmax + IoU suppression.
// ---------------------------------------------------------------------------
__global__ __launch_bounds__(256) void nms_kernel(const float* __restrict__ boxes,
                                                  float* __restrict__ scores,
                                                  float* __restrict__ dout)
{
  __shared__ float sval[256];
  __shared__ int   sidx[256];
  __shared__ float selb[4];
  __shared__ float selarea;
  __shared__ int   selok;

  const int n   = blockIdx.x;
  const int tid = threadIdx.x;
  const float* bx = boxes + (size_t)n * KA * 4;
  float* sc   = scores + (size_t)n * KA;
  float* rois = dout + ROIS_OFF + (size_t)n * POSTN * 4;
  int*   ridx = (int*)dout + RIDX_OFF + (size_t)n * POSTN;
  const float NEGINF = -__builtin_inff();

  for (int it = 0; it < POSTN; ++it) {
    float bv = NEGINF;
    int bi = -1;
    for (int k = tid; k < KA; k += 256) {
      float v = sc[k];
      if (v > bv) { bv = v; bi = k; }
    }
    sval[tid] = bv;
    sidx[tid] = bi;
    __syncthreads();
    for (int s = 128; s > 0; s >>= 1) {
      if (tid < s && sval[tid + s] > sval[tid]) {
        sval[tid] = sval[tid + s];
        sidx[tid] = sidx[tid + s];
      }
      __syncthreads();
    }
    if (tid == 0) {
      int i0 = sidx[0];
      float v0 = sval[0];
      int ok = (i0 >= 0) && (v0 > NEGINF);
      float b0 = 0.0f, b1 = 0.0f, b2 = 0.0f, b3 = 0.0f;
      if (ok) {
        b0 = bx[(size_t)i0 * 4 + 0];
        b1 = bx[(size_t)i0 * 4 + 1];
        b2 = bx[(size_t)i0 * 4 + 2];
        b3 = bx[(size_t)i0 * 4 + 3];
        sc[i0] = NEGINF;
      }
      selok = ok;
      selb[0] = b0; selb[1] = b1; selb[2] = b2; selb[3] = b3;
      selarea = (b2 - b0) * (b3 - b1);
      rois[it * 4 + 0] = b0;
      rois[it * 4 + 1] = b1;
      rois[it * 4 + 2] = b2;
      rois[it * 4 + 3] = b3;
      ridx[it] = n;
    }
    __syncthreads();
    if (selok) {
      float y1 = selb[0], x1 = selb[1], y2 = selb[2], x2 = selb[3], ar = selarea;
      for (int k = tid; k < KA; k += 256) {
        float v = sc[k];
        if (v > NEGINF) {
          float ky1 = bx[(size_t)k * 4 + 0];
          float kx1 = bx[(size_t)k * 4 + 1];
          float ky2 = bx[(size_t)k * 4 + 2];
          float kx2 = bx[(size_t)k * 4 + 3];
          float yy1 = fmaxf(y1, ky1), xx1 = fmaxf(x1, kx1);
          float yy2 = fminf(y2, ky2), xx2 = fminf(x2, kx2);
          float inter = fmaxf(yy2 - yy1, 0.0f) * fmaxf(xx2 - xx1, 0.0f);
          float ka = (ky2 - ky1) * (kx2 - kx1);
          float iou = inter / (ka + ar - inter + 1e-9f);
          if (iou >= 0.7f) sc[k] = NEGINF;
        }
      }
    }
    __syncthreads();
  }
}

// ---------------------------------------------------------------------------
// Launch
// ---------------------------------------------------------------------------
extern "C" void kernel_launch(void* const* d_in, const int* in_sizes, int n_in,
                              void* d_out, int out_size, void* d_ws, size_t ws_size,
                              hipStream_t stream) {
  (void)in_sizes; (void)n_in; (void)out_size; (void)ws_size;
  const float* x       = (const float*)d_in[0];
  const float* conv1_w = (const float*)d_in[1];
  const float* conv1_b = (const float*)d_in[2];
  const float* score_w = (const float*)d_in[3];
  const float* score_b = (const float*)d_in[4];
  const float* loc_w   = (const float*)d_in[5];
  const float* loc_b   = (const float*)d_in[6];
  const int*   img_h   = (const int*)d_in[7];
  const int*   img_w   = (const int*)d_in[8];

  char* ws = (char*)d_ws;
  __bf16* xb     = (__bf16*)(ws + 0);            // 33,554,432 B
  __bf16* wt     = (__bf16*)(ws + 33554432);     //  1,179,648 B
  __bf16* w2t    = (__bf16*)(ws + 34734080);     //     32,768 B
  float*  b2     = (float*) (ws + 34766848);     //        512 B
  __bf16* hbuf   = (__bf16*)(ws + 34767360);     // 33,554,432 B
  float*  out2   = (float*) (ws + 68321792);     // 16,777,216 B
  float*  scores = (float*) (ws + 85099008);     //  2,359,296 B
  float*  boxes  = (float*) (ws + 87458304);     //  9,437,184 B
  float*  out    = (float*)d_out;

  prep_x <<<(NIMG * CIN * HDIM * WDIM) / 256, 256, 0, stream>>>(x, xb);
  prep_w1<<<(MIDCH * KCONV + 255) / 256, 256, 0, stream>>>(conv1_w, wt);
  prep_w2<<<(N2 * 256) / 256, 256, 0, stream>>>(loc_w, score_w, loc_b, score_b, w2t, b2);

  conv3x3_wmma<<<dim3(MROWS / 128, MIDCH / 64), 256, 0, stream>>>(xb, wt, conv1_b, hbuf);
  gemm2_wmma  <<<dim3(MROWS / 128), 256, 0, stream>>>(hbuf, w2t, b2, out2);

  decode_kernel<<<(NIMG * KA + 255) / 256, 256, 0, stream>>>(out2, img_h, img_w,
                                                             out, scores, boxes);
  nms_kernel<<<NIMG, 256, 0, stream>>>(boxes, scores, out);
}
